// TemporalGATNet_35055523070303
// MI455X (gfx1250) — matrix operand order repaired and checked by reading, not compiled, not run on hardware
//
#include <hip/hip_runtime.h>
#include <hip/hip_bf16.h>

// ---------------- problem constants (from reference) ----------------
#define N_NODES   33
#define SEQLEN    24
#define BATCH     64
#define GRAPHS    (BATCH * SEQLEN)          // 1536
#define EPG       64
#define NEDGES    (GRAPHS * EPG)            // 98304
#define NNODES_T  (GRAPHS * N_NODES)        // 50688
#define IN_CH     2
#define HID       64
#define HEADS     8
#define HC        (HID * HEADS)             // 512
#define EDGE_DIM  3
#define L1H       128
#define L2H       64
#define NCLS      4
#define NEG_SLOPE 0.2f
#define FEAT      (N_NODES * HID)           // 2112

typedef _Float16 half_t;
typedef __attribute__((ext_vector_type(16))) _Float16 v16h;
typedef __attribute__((ext_vector_type(8)))  float    v8f;

// ---------------- small helpers ----------------
__device__ __forceinline__ float sigmoidf_(float x) { return 1.0f / (1.0f + __expf(-x)); }
__device__ __forceinline__ float eluf_(float x)     { return x > 0.0f ? x : (__expf(x) - 1.0f); }
__device__ __forceinline__ float lreluf_(float x)   { return x > 0.0f ? x : NEG_SLOPE * x; }

__device__ void atomicMaxF(float* addr, float v) {
    unsigned int* ua = (unsigned int*)addr;
    unsigned int cur = __float_as_uint(*addr);
    while (__uint_as_float(cur) < v) {
        unsigned int prev = atomicCAS(ua, cur, __float_as_uint(v));
        if (prev == cur) break;
        cur = prev;
    }
}

// ---------------- generic fill ----------------
__global__ void fill_f32(float* p, float v, int n) {
    int i = blockIdx.x * blockDim.x + threadIdx.x;
    if (i < n) p[i] = v;
}

// ---------------- tiny-K GEMM (GAT layer 1 projections), f16 out ----------------
__global__ void smallk_gemm_f16(const float* __restrict__ A, const float* __restrict__ W,
                                half_t* __restrict__ out, int M, int K, int Nc) {
    int idx = blockIdx.x * blockDim.x + threadIdx.x;
    if (idx >= M * Nc) return;
    int m = idx / Nc, n = idx % Nc;
    float s = 0.0f;
    for (int k = 0; k < K; ++k) s += A[(size_t)m * K + k] * W[k * Nc + n];
    out[idx] = (half_t)s;
}

// ---------------- weight conversion into WMMA-fragment-packed f16 ----------------
// Packed layout: for k-slab kt (32 k's) and column n, the 32 halves
// Bp[((kt*Nc + n)*32 + kk)] = B[kt*32+kk][n], kk ascending.  A lane with
// halfSel h reads 16 contiguous halves at offset h*16 -> 2x global_load_b128.
__global__ void conv_pack_f16(const float* __restrict__ in /* K x Nc row-major */,
                              half_t* __restrict__ out, int K, int Nc) {
    int i = blockIdx.x * blockDim.x + threadIdx.x;
    if (i >= K * Nc) return;
    int kk = i & 31;
    int rest = i >> 5;
    int n  = rest % Nc;
    int kt = rest / Nc;
    out[i] = (half_t)in[(size_t)(kt * 32 + kk) * Nc + n];
}
// in: (R rows x K cols) row-major, logical B[k][n] = in[n][k]  (R == Nc)
__global__ void conv_t_pack_f16(const float* __restrict__ in,
                                half_t* __restrict__ out, int R, int K) {
    int i = blockIdx.x * blockDim.x + threadIdx.x;
    if (i >= R * K) return;
    int kk = i & 31;
    int rest = i >> 5;
    int n  = rest % R;
    int kt = rest / R;
    out[i] = (half_t)in[(size_t)n * K + kt * 32 + kk];   // contiguous along kk
}

// ---------------- GATv2 edge kernels (shared by both layers) ----------------
__global__ void edge_logits(const half_t* __restrict__ xl, const half_t* __restrict__ xr,
                            const float* __restrict__ eattr, const float* __restrict__ We,
                            const float* __restrict__ att,
                            const int* __restrict__ src, const int* __restrict__ dst,
                            float* __restrict__ logits, float* __restrict__ mx) {
    int idx = blockIdx.x * blockDim.x + threadIdx.x;
    if (idx >= NEDGES * HEADS) return;
    int e = idx >> 3, h = idx & 7;
    int s = src[e], d = dst[e];
    float a0 = eattr[e * 3], a1 = eattr[e * 3 + 1], a2 = eattr[e * 3 + 2];
    const half_t* xls = xl + (size_t)s * HC + h * HID;
    const half_t* xrd = xr + (size_t)d * HC + h * HID;
    const float* ath  = att + h * HID;
    float acc = 0.0f;
    #pragma unroll 4
    for (int c = 0; c < HID; ++c) {
        int hc = h * HID + c;
        float em = a0 * We[hc] + a1 * We[HC + hc] + a2 * We[2 * HC + hc];
        float m  = (float)xls[c] + (float)xrd[c] + em;
        acc += ath[c] * lreluf_(m);
    }
    logits[idx] = acc;
    atomicMaxF(&mx[d * HEADS + h], acc);
}

__global__ void edge_exp(float* __restrict__ logits, const float* __restrict__ mx,
                         const int* __restrict__ dst, float* __restrict__ den) {
    int idx = blockIdx.x * blockDim.x + threadIdx.x;
    if (idx >= NEDGES * HEADS) return;
    int e = idx >> 3, h = idx & 7;
    int d = dst[e];
    float ex = __expf(logits[idx] - mx[d * HEADS + h]);
    logits[idx] = ex;                         // in-place: logits now holds ex
    atomicAdd(&den[d * HEADS + h], ex);
}

__global__ void edge_scatter(const float* __restrict__ ex, const float* __restrict__ den,
                             const half_t* __restrict__ xl,
                             const int* __restrict__ src, const int* __restrict__ dst,
                             float* __restrict__ acc) {
    int idx = blockIdx.x * blockDim.x + threadIdx.x;
    if (idx >= NEDGES * HC) return;
    int e  = idx >> 9;          // /512
    int hc = idx & (HC - 1);
    int h  = hc >> 6;
    int s = src[e], d = dst[e];
    float alpha = ex[e * HEADS + h] / (den[d * HEADS + h] + 1e-16f);
    atomicAdd(&acc[(size_t)d * HC + hc], alpha * (float)xl[(size_t)s * HC + hc]);
}

// ---------------- post-aggregation epilogues ----------------
__global__ void bias_elu_f16(const float* __restrict__ acc, const float* __restrict__ b,
                             half_t* __restrict__ out) {
    int idx = blockIdx.x * blockDim.x + threadIdx.x;
    if (idx >= NNODES_T * HC) return;
    int c = idx & (HC - 1);
    out[idx] = (half_t)eluf_(acc[idx] + b[c]);
}

__global__ void mean_bias_elu_f16(const float* __restrict__ acc, const float* __restrict__ b,
                                  half_t* __restrict__ out) {
    int idx = blockIdx.x * blockDim.x + threadIdx.x;
    if (idx >= NNODES_T * HID) return;
    int n = idx >> 6, c = idx & (HID - 1);
    float s = 0.0f;
    #pragma unroll
    for (int h = 0; h < HEADS; ++h) s += acc[(size_t)n * HC + h * HID + c];
    out[idx] = (half_t)eluf_(s * (1.0f / HEADS) + b[c]);
}

// ---------------- WMMA GEMM: C(MxNc) = A(MxK,f16) * Bp(packed f16) ----------------
// Block = 256 threads = 8 waves.  Block covers 16 M rows x 256 N columns;
// each wave owns two 16x16 tiles (A fragment reused, 2 accumulators).
// A k-slab (16x32 halves) staged in LDS with 8-half chunks in k order {0,1,2,3}
// so a lane's full 16-half fragment is 32 contiguous LDS bytes (2x ds_load_b128).
// B is pre-packed so a lane's fragment is 32 contiguous global bytes.
// Fragment layouts per CDNA5 ISA 7.12.2 (wave32): 16-bit A 16x32, f32 C/D in 8
// VGPRs with lanes 0-15 -> M=r, lanes 16-31 -> M=r+8.
__global__ __launch_bounds__(256) void wmma_gemm(const half_t* __restrict__ A,
                                                 const half_t* __restrict__ Bp,
                                                 void* __restrict__ Cout,
                                                 int M, int K, int Nc,
                                                 const float* __restrict__ bias0,
                                                 const float* __restrict__ bias1,
                                                 int storeHalf) {
    __shared__ alignas(32) half_t Alds[16 * 32];
    const int nGroups = Nc >> 8;                       // 256-col groups
    const int mTile   = blockIdx.x / nGroups;
    const int nBase   = (blockIdx.x % nGroups) << 8;
    const int tid     = threadIdx.x;
    const int wave    = tid >> 5;
    const int lane    = tid & 31;
    const int halfSel = lane >> 4;
    const int mrow    = lane & 15;
    const int nCol0   = nBase + (wave << 5) + (lane & 15);
    const int nCol1   = nCol0 + 16;

    v8f c0 = {}, c1 = {};

    #pragma unroll 2
    for (int k0 = 0; k0 < K; k0 += 32) {
        __syncthreads();
        if (tid < 64) {   // 64 threads stage 16x32 halves of A via b128 copies
            int r  = tid >> 2;
            int ch = tid & 3;
            int pc = ch;                               // chunk order 0,1,2,3 (contiguous k)
            const uint4 v = *(const uint4*)(A + (size_t)(mTile * 16 + r) * K + k0 + ch * 8);
            *(uint4*)(Alds + r * 32 + pc * 8) = v;
        }
        __syncthreads();

        const int kt = k0 >> 5;
        if (k0 + 32 < K)   // pull next B k-slab toward cache (global_prefetch_b8)
            __builtin_prefetch(Bp + ((size_t)(kt + 1) * Nc + nCol0) * 32, 0, 1);

        v16h af = *(const v16h*)(Alds + mrow * 32 + halfSel * 16);
        v16h b0 = *(const v16h*)(Bp + ((size_t)kt * Nc + nCol0) * 32 + halfSel * 16);
        v16h b1 = *(const v16h*)(Bp + ((size_t)kt * Nc + nCol1) * 32 + halfSel * 16);

        c0 = __builtin_amdgcn_wmma_f32_16x16x32_f16(false, af, false, b0,
                                                    (short)0, c0, false, false);
        c1 = __builtin_amdgcn_wmma_f32_16x16x32_f16(false, af, false, b1,
                                                    (short)0, c1, false, false);
    }

    #pragma unroll
    for (int r = 0; r < 8; ++r) {
        int row = mTile * 16 + r + 8 * halfSel;
        float v0 = c0[r], v1 = c1[r];
        if (bias0) { v0 += bias0[nCol0]; v1 += bias0[nCol1]; }
        if (bias1) { v0 += bias1[nCol0]; v1 += bias1[nCol1]; }
        if (storeHalf) {
            ((half_t*)Cout)[(size_t)row * Nc + nCol0] = (half_t)v0;
            ((half_t*)Cout)[(size_t)row * Nc + nCol1] = (half_t)v1;
        } else {
            ((float*)Cout)[(size_t)row * Nc + nCol0] = v0;
            ((float*)Cout)[(size_t)row * Nc + nCol1] = v1;
        }
    }
}

// ---------------- persistent LSTM scan (single workgroup; h in LDS, c in VGPRs) ----------------
template <int B, int H, int T, int NT>
__global__ __launch_bounds__(NT) void lstm_scan(const float* __restrict__ Xg,   // (B*T) x 4H, row = b*T+t
                                                const float* __restrict__ Whh,  // 4H x H
                                                half_t* __restrict__ ysOut,     // (B*T) x H  (or null)
                                                float* __restrict__ hlast) {    // B x H      (or null)
    __shared__ float hs[B * H];
    constexpr int CELLS = B * H;
    constexpr int CPT   = CELLS / NT;
    const int tid = threadIdx.x;
    float creg[CPT], hn[CPT];
    #pragma unroll
    for (int k = 0; k < CPT; ++k) { creg[k] = 0.0f; hs[tid + k * NT] = 0.0f; }
    __syncthreads();

    for (int t = 0; t < T; ++t) {
        for (int k = 0; k < CPT; ++k) {
            int cid = tid + k * NT;
            int b = cid / H, j = cid % H;
            const float* xg = Xg + (size_t)(b * T + t) * (4 * H);
            float gi = xg[j], gf = xg[H + j], gg = xg[2 * H + j], go = xg[3 * H + j];
            const float* wI = Whh + (size_t)j * H;
            const float* wF = Whh + (size_t)(H + j) * H;
            const float* wG = Whh + (size_t)(2 * H + j) * H;
            const float* wO = Whh + (size_t)(3 * H + j) * H;
            const float* hb = hs + b * H;
            for (int i = 0; i < H; ++i) {
                float hv = hb[i];
                gi += hv * wI[i]; gf += hv * wF[i]; gg += hv * wG[i]; go += hv * wO[i];
            }
            gi = sigmoidf_(gi); gf = sigmoidf_(gf); gg = tanhf(gg); go = sigmoidf_(go);
            float cc = gf * creg[k] + gi * gg;
            creg[k] = cc;
            hn[k] = go * tanhf(cc);
        }
        __syncthreads();
        for (int k = 0; k < CPT; ++k) {
            int cid = tid + k * NT;
            int b = cid / H, j = cid % H;
            hs[cid] = hn[k];
            if (ysOut) ysOut[(size_t)(b * T + t) * H + j] = (half_t)hn[k];
            if (hlast && t == T - 1) hlast[cid] = hn[k];
        }
        __syncthreads();
    }
}

// ---------------- final FC ----------------
__global__ void fc_kernel(const float* __restrict__ h, const float* __restrict__ W,
                          const float* __restrict__ b, float* __restrict__ out) {
    int idx = blockIdx.x * blockDim.x + threadIdx.x;
    if (idx >= BATCH * NCLS) return;
    int bb = idx >> 2, c = idx & 3;
    float s = b[c];
    #pragma unroll 8
    for (int j = 0; j < L2H; ++j) s += h[bb * L2H + j] * W[c * L2H + j];
    out[idx] = s;
}

// =================================================================
static inline size_t al256(size_t x) { return (x + 255) & ~(size_t)255; }
static inline int gridOf(long long n, int bs) { return (int)((n + bs - 1) / bs); }

extern "C" void kernel_launch(void* const* d_in, const int* in_sizes, int n_in,
                              void* d_out, int out_size, void* d_ws, size_t ws_size,
                              hipStream_t stream) {
    const float* x      = (const float*)d_in[0];
    const int*   eidx   = (const int*)  d_in[1];
    const float* eattr  = (const float*)d_in[2];
    const float* g1_Wl  = (const float*)d_in[3];
    const float* g1_Wr  = (const float*)d_in[4];
    const float* g1_We  = (const float*)d_in[5];
    const float* g1_att = (const float*)d_in[6];
    const float* g1_b   = (const float*)d_in[7];
    const float* g2_Wl  = (const float*)d_in[8];
    const float* g2_Wr  = (const float*)d_in[9];
    const float* g2_We  = (const float*)d_in[10];
    const float* g2_att = (const float*)d_in[11];
    const float* g2_b   = (const float*)d_in[12];
    const float* l1_Wih = (const float*)d_in[13];
    const float* l1_Whh = (const float*)d_in[14];
    const float* l1_bih = (const float*)d_in[15];
    const float* l1_bhh = (const float*)d_in[16];
    const float* l2_Wih = (const float*)d_in[17];
    const float* l2_Whh = (const float*)d_in[18];
    const float* l2_bih = (const float*)d_in[19];
    const float* l2_bhh = (const float*)d_in[20];
    const float* fc_W   = (const float*)d_in[21];
    const float* fc_b   = (const float*)d_in[22];
    float* out = (float*)d_out;

    const int* src = eidx;
    const int* dst = eidx + NEDGES;

    // ---------------- workspace layout ----------------
    char* p = (char*)d_ws;
    size_t off = 0;
    auto take = [&](size_t bytes) { void* r = p + off; off += al256(bytes); return r; };
    half_t* XLH  = (half_t*)take((size_t)NNODES_T * HC * 2);   // xl (f16), both layers
    half_t* XRH  = (half_t*)take((size_t)NNODES_T * HC * 2);   // xr (f16)
    float*  ACC  = (float*) take((size_t)NNODES_T * HC * 4);   // scatter accumulator
    half_t* H1H  = (half_t*)take((size_t)NNODES_T * HC * 2);   // h1 (f16) = GAT2 input
    float*  LOG  = (float*) take((size_t)NEDGES * HEADS * 4);  // logits -> ex (in place)
    float*  MX   = (float*) take((size_t)NNODES_T * HEADS * 4);
    float*  DEN  = (float*) take((size_t)NNODES_T * HEADS * 4);
    half_t* BH   = (half_t*)take((size_t)FEAT * HC * 2);       // packed f16 weights (max l1_Wih^T)
    half_t* SEQH = (half_t*)take((size_t)GRAPHS * FEAT * 2);   // LSTM1 input seq (f16)
    float*  XG1  = (float*) take((size_t)GRAPHS * 4 * L1H * 4);
    half_t* YS1H = (half_t*)take((size_t)GRAPHS * L1H * 2);
    float*  XG2  = (float*) take((size_t)GRAPHS * 4 * L2H * 4);
    float*  HLAST= (float*) take((size_t)BATCH * L2H * 4);

    const int BS = 256;
    const long long nNodeHC = (long long)NNODES_T * HC;
    const long long nEdgeH  = (long long)NEDGES * HEADS;
    const long long nEdgeHC = (long long)NEDGES * HC;

    // ============ GAT layer 1 ============
    smallk_gemm_f16<<<gridOf(nNodeHC, BS), BS, 0, stream>>>(x, g1_Wl, XLH, NNODES_T, IN_CH, HC);
    smallk_gemm_f16<<<gridOf(nNodeHC, BS), BS, 0, stream>>>(x, g1_Wr, XRH, NNODES_T, IN_CH, HC);
    fill_f32<<<gridOf(NNODES_T * HEADS, BS), BS, 0, stream>>>(MX, -1e30f, NNODES_T * HEADS);
    fill_f32<<<gridOf(NNODES_T * HEADS, BS), BS, 0, stream>>>(DEN, 0.0f, NNODES_T * HEADS);
    fill_f32<<<gridOf(nNodeHC, BS), BS, 0, stream>>>(ACC, 0.0f, (int)nNodeHC);
    edge_logits<<<gridOf(nEdgeH, BS), BS, 0, stream>>>(XLH, XRH, eattr, g1_We, g1_att, src, dst, LOG, MX);
    edge_exp<<<gridOf(nEdgeH, BS), BS, 0, stream>>>(LOG, MX, dst, DEN);
    edge_scatter<<<gridOf(nEdgeHC, BS), BS, 0, stream>>>(LOG, DEN, XLH, src, dst, ACC);
    bias_elu_f16<<<gridOf(nNodeHC, BS), BS, 0, stream>>>(ACC, g1_b, H1H);

    // ============ GAT layer 2 (WMMA GEMMs) ============
    conv_pack_f16<<<gridOf(HC * HC, BS), BS, 0, stream>>>(g2_Wl, BH, HC, HC);
    wmma_gemm<<<(NNODES_T / 16) * (HC / 256), 256, 0, stream>>>(H1H, BH, XLH, NNODES_T, HC, HC,
                                                                nullptr, nullptr, 1);
    conv_pack_f16<<<gridOf(HC * HC, BS), BS, 0, stream>>>(g2_Wr, BH, HC, HC);
    wmma_gemm<<<(NNODES_T / 16) * (HC / 256), 256, 0, stream>>>(H1H, BH, XRH, NNODES_T, HC, HC,
                                                                nullptr, nullptr, 1);
    fill_f32<<<gridOf(NNODES_T * HEADS, BS), BS, 0, stream>>>(MX, -1e30f, NNODES_T * HEADS);
    fill_f32<<<gridOf(NNODES_T * HEADS, BS), BS, 0, stream>>>(DEN, 0.0f, NNODES_T * HEADS);
    fill_f32<<<gridOf(nNodeHC, BS), BS, 0, stream>>>(ACC, 0.0f, (int)nNodeHC);
    edge_logits<<<gridOf(nEdgeH, BS), BS, 0, stream>>>(XLH, XRH, eattr, g2_We, g2_att, src, dst, LOG, MX);
    edge_exp<<<gridOf(nEdgeH, BS), BS, 0, stream>>>(LOG, MX, dst, DEN);
    edge_scatter<<<gridOf(nEdgeHC, BS), BS, 0, stream>>>(LOG, DEN, XLH, src, dst, ACC);
    mean_bias_elu_f16<<<gridOf((long long)NNODES_T * HID, BS), BS, 0, stream>>>(ACC, g2_b, SEQH);

    // ============ LSTM 1 ============
    conv_t_pack_f16<<<gridOf(4 * L1H * FEAT, BS), BS, 0, stream>>>(l1_Wih, BH, 4 * L1H, FEAT);
    wmma_gemm<<<(GRAPHS / 16) * ((4 * L1H) / 256), 256, 0, stream>>>(SEQH, BH, XG1, GRAPHS, FEAT, 4 * L1H,
                                                                     l1_bih, l1_bhh, 0);
    lstm_scan<BATCH, L1H, SEQLEN, 256><<<1, 256, 0, stream>>>(XG1, l1_Whh, YS1H, nullptr);

    // ============ LSTM 2 ============
    conv_t_pack_f16<<<gridOf(4 * L2H * L1H, BS), BS, 0, stream>>>(l2_Wih, BH, 4 * L2H, L1H);
    wmma_gemm<<<(GRAPHS / 16) * ((4 * L2H) / 256), 256, 0, stream>>>(YS1H, BH, XG2, GRAPHS, L1H, 4 * L2H,
                                                                     l2_bih, l2_bhh, 0);
    lstm_scan<BATCH, L2H, SEQLEN, 256><<<1, 256, 0, stream>>>(XG2, l2_Whh, nullptr, HLAST);

    // ============ FC head ============
    fc_kernel<<<1, 256, 0, stream>>>(HLAST, fc_W, fc_b, out);

    (void)in_sizes; (void)n_in; (void)out_size; (void)ws_size;
}
